// LaneATT_33036888441568
// MI455X (gfx1250) — matrix operand order repaired and checked by reading, not compiled, not run on hardware
//
#include <hip/hip_runtime.h>
#include <hip/hip_bf16.h>
#include <math.h>

// ---------------------------------------------------------------------------
// LaneATT on MI455X (gfx1250, wave32): bf16 WMMA 16x16x32 GEMMs with fp32
// accumulate, TDM (tensor_load_to_lds) double-buffered weight staging, and an
// LDS-resident O(n^2) NMS.
//
// Only the 42 edge pixels (col0, col19, row10) of the 11x20 feature map are
// ever consumed by the proposal gather -> M = 32*42 = 1344 GEMM rows.
// d_out = proposals (32*714*77) ++ keep (32*714) ++ scores (32*714), all f32.
// ---------------------------------------------------------------------------

#define B_BATCH   32
#define C_IN      512
#define C_MID     1024
#define FH        11
#define FW        20
#define N_EDGE    42                      // 11 left + 11 right + 20 bottom
#define M_ROWS    (B_BATCH * N_EDGE)      // 1344
#define M_TILES   (M_ROWS / 16)           // 84
#define N_ANG     17
#define N_OFF     72
#define N_PRED    77
#define N_PROP    (N_ANG * N_EDGE)        // 714
#define N_CLS     (2 * N_ANG)             // 34
#define N_REG     (N_ANG * (N_OFF + 1))   // 1241
#define N_HEAD    (N_CLS + N_REG)         // 1275
#define N_HEAD_P  1280                    // padded
#define NMS_THR   15.0f
#define S2_WAVES  6                       // M-tiles per stage-2 block

typedef __attribute__((ext_vector_type(16))) __bf16       v16bf;
typedef __attribute__((ext_vector_type(8)))  float        v8f;
typedef __attribute__((ext_vector_type(4)))  unsigned int u32x4;
typedef __attribute__((ext_vector_type(8)))  int          i32x8;
typedef __attribute__((ext_vector_type(4)))  int          i32x4;

#if __has_builtin(__builtin_amdgcn_tensor_load_to_lds)
#define HAVE_TDM 1
#else
#define HAVE_TDM 0
#endif

__device__ __forceinline__ void edge_to_hw(int e, int& h, int& w) {
    if (e < FH)            { h = e;        w = 0;      }
    else if (e < 2 * FH)   { h = e - FH;   w = FW - 1; }
    else                   { h = FH - 1;   w = e - 2 * FH; }
}

#if HAVE_TDM
// TDM: async 2D bf16 tile (tile_k x tile_rows) from a row-major
// [tensor_rows x tensor_k] bf16 tensor into LDS.  D# packed per ISA 8.3/8.4.
// 6-arg builtin form (this toolchain): (g0, g1, g2, g3, g4, cpol).
__device__ __forceinline__ void tdm_load_2d_bf16(unsigned lds_off,
                                                 const __bf16* gsrc,
                                                 int tile_k, int tile_rows,
                                                 int tensor_k, int tensor_rows) {
    unsigned long long ga = (unsigned long long)(size_t)gsrc;
    u32x4 g0;
    g0[0] = 1u;                                            // count=1, user desc
    g0[1] = lds_off;                                       // lds_addr (bytes)
    g0[2] = (unsigned)ga;                                  // global_addr[31:0]
    g0[3] = (unsigned)((ga >> 32) & 0x01ffffffu) | (2u << 30); // addr[56:32] | type=2
    i32x8 g1;
    g1[0] = (1 << 16);                                     // data_size=1 -> 2 bytes
    g1[1] = (int)(((unsigned)tensor_k & 0xffffu) << 16);   // tensor_dim0[15:0]
    g1[2] = (int)((((unsigned)tensor_k >> 16) & 0xffffu)   // tensor_dim0[31:16]
                  | (((unsigned)tensor_rows & 0xffffu) << 16)); // tensor_dim1[15:0]
    g1[3] = (int)((((unsigned)tensor_rows >> 16) & 0xffffu)
                  | (((unsigned)tile_k & 0xffffu) << 16)); // tile_dim0
    g1[4] = (int)((unsigned)tile_rows & 0xffffu);          // tile_dim1 (tile_dim2=0)
    g1[5] = tensor_k;                                      // tensor_dim0_stride[31:0]
    g1[6] = 0;                                             // stride0[47:32], stride1[15:0]
    g1[7] = 0;
    i32x4 gz4 = {0, 0, 0, 0};
    i32x8 gz8 = {0, 0, 0, 0, 0, 0, 0, 0};
    __builtin_amdgcn_tensor_load_to_lds(g0, g1, gz4, gz4, gz8, 0);
}
#endif

// ---------------------------------------------------------------------------
// Weight packing: one-shot f32 -> bf16 conversion.
//   Wcb : (1024, 512) bf16 copy of W_conv
//   Whb : (1280, 1024) bf16 fused [W_cls ; W_reg ; zeros] head weights
// ---------------------------------------------------------------------------
__global__ void laneatt_pack_weights(const float* __restrict__ Wconv,
                                     const float* __restrict__ Wcls,
                                     const float* __restrict__ Wreg,
                                     __bf16* __restrict__ Wcb,
                                     __bf16* __restrict__ Whb) {
    const int n_h = N_HEAD_P * C_MID;                 // 1,310,720
    const int n_c = C_MID * C_IN;                     //   524,288
    const int gid = blockIdx.x * blockDim.x + threadIdx.x;
    if (gid < n_h) {
        const int o = gid / C_MID, k = gid - o * C_MID;
        float v = 0.0f;
        if (o < N_CLS)       v = Wcls[(size_t)o * C_MID + k];
        else if (o < N_HEAD) v = Wreg[(size_t)(o - N_CLS) * C_MID + k];
        Whb[gid] = (__bf16)v;
    } else if (gid < n_h + n_c) {
        const int idx = gid - n_h;
        Wcb[idx] = (__bf16)Wconv[idx];
    }
}

// ---------------------------------------------------------------------------
// Stage 1: features[m, o] = feat_edge[m, :] . W_conv[o, :] + b_conv[o]
//   M=1344, K=512, N=1024.  One wave -> 16x64 tile (4 accumulators):
//   the strided A gather (stride 220 f32 + cvt) is amortized over 4 WMMAs.
// ---------------------------------------------------------------------------
__global__ void laneatt_stage1_conv(const float* __restrict__ feat,
                                    const __bf16* __restrict__ Wcb,
                                    const float* __restrict__ bc,
                                    __bf16* __restrict__ featb) {
    const int wave = (blockIdx.x * blockDim.x + threadIdx.x) >> 5;
    const int lane = threadIdx.x & 31;
    const int nGrp = C_MID / 64;                  // 16 column groups
    const int mt = wave / nGrp;
    const int ng = wave - mt * nGrp;
    if (mt >= M_TILES) return;                    // wave-uniform exit

    const int row   = lane & 15;                  // A: M row
    const int khalf = lane >> 4;                  // K half owned by this lane

    const int m = mt * 16 + row;
    const int b = m / N_EDGE, e = m - b * N_EDGE;
    int h, w; edge_to_hw(e, h, w);
    const float* abase = feat + (size_t)b * C_IN * (FH * FW) + h * FW + w;

    v8f acc[4] = {};
    for (int k0 = 0; k0 < C_IN; k0 += 32) {
        v16bf A;
        #pragma unroll
        for (int t = 0; t < 8; ++t) {
            A[t]     = (__bf16)abase[(size_t)(k0 + khalf * 16 +     t) * (FH * FW)];
            A[t + 8] = (__bf16)abase[(size_t)(k0 + khalf * 16 + 8 + t) * (FH * FW)];
        }
        #pragma unroll
        for (int t = 0; t < 4; ++t) {
            const v16bf Bv = *(const v16bf*)(Wcb
                + (size_t)(ng * 64 + t * 16 + (lane & 15)) * C_IN + k0 + khalf * 16);
            acc[t] = __builtin_amdgcn_wmma_f32_16x16x32_bf16(
                         false, A, false, Bv, (short)0, acc[t], false, false);
        }
    }

    #pragma unroll
    for (int t = 0; t < 4; ++t) {
        const int ncol = ng * 64 + t * 16 + (lane & 15);
        const float bias = bc[ncol];
        #pragma unroll
        for (int r = 0; r < 8; ++r) {
            const int mm = mt * 16 + khalf * 8 + r;
            featb[(size_t)mm * C_MID + ncol] = (__bf16)(acc[t][r] + bias);
        }
    }
}

// ---------------------------------------------------------------------------
// Stage 2: fused head GEMM  head[m, o] = features[m, :] . Whb[o, :] + bias
//   M=1344, K=1024, N=1280.  Block = 6 waves x (16 rows each) sharing one
//   64-col group.  The 64x32 bf16 B tile is TDM-staged into LDS, double
//   buffered: wave 0 issues tensor_load_to_lds for tile i+1, waits
//   s_wait_tensorcnt<=1 for tile i, then the workgroup barrier publishes it.
// ---------------------------------------------------------------------------
__global__ void __launch_bounds__(32 * S2_WAVES)
laneatt_stage2_heads(const __bf16* __restrict__ featb,
                     const __bf16* __restrict__ Whb,
                     const float* __restrict__ bcls,
                     const float* __restrict__ breg,
                     float* __restrict__ head) {
    __shared__ __align__(128) __bf16 btile[2][64 * 32];

    const int ng   = blockIdx.x;                  // 0..19  (64-col group)
    const int mg   = blockIdx.y;                  // 0..13  (6 M-tiles)
    const int wv   = threadIdx.x >> 5;
    const int lane = threadIdx.x & 31;
    const int mt   = mg * S2_WAVES + wv;          // < 84 always

    const int row   = lane & 15;
    const int khalf = lane >> 4;

    const __bf16* abase = featb + (size_t)(mt * 16 + row) * C_MID;
    const __bf16* wbase = Whb + (size_t)(ng * 64) * C_MID;
    const int nK = C_MID / 32;                    // 32 K-steps

    auto stage_issue = [&](int i) {               // tile i -> btile[i&1]
#if HAVE_TDM
        if (threadIdx.x < 32)                     // one TDM issue per block
            tdm_load_2d_bf16((unsigned)(size_t)(void*)&btile[i & 1][0],
                             wbase + i * 32, 32, 64, C_MID, N_HEAD_P);
#else
        for (int t = threadIdx.x; t < 64 * 32; t += blockDim.x)
            btile[i & 1][t] = wbase[(size_t)(t >> 5) * C_MID + i * 32 + (t & 31)];
#endif
    };

    stage_issue(0);
    v8f acc[4] = {};
    for (int i = 0; i < nK; ++i) {
        const int k0 = i * 32;
        if (i + 1 < nK) {                         // prefetch next tile
            stage_issue(i + 1);
#if HAVE_TDM
            if (threadIdx.x < 32)                 // tile i landed, i+1 in flight
                __builtin_amdgcn_s_wait_tensorcnt(1);
#endif
        } else {
#if HAVE_TDM
            if (threadIdx.x < 32)                 // last tile: drain
                __builtin_amdgcn_s_wait_tensorcnt(0);
#endif
        }
        __syncthreads();

        v16bf A;
        #pragma unroll
        for (int t = 0; t < 8; ++t) {             // contiguous bf16 -> b128 loads
            A[t]     = abase[k0 + khalf * 16 +     t];
            A[t + 8] = abase[k0 + khalf * 16 + 8 + t];
        }
        const __bf16* bb = &btile[i & 1][0];
        #pragma unroll
        for (int t = 0; t < 4; ++t) {
            const v16bf Bv = *(const v16bf*)(bb + (t * 16 + (lane & 15)) * 32 + khalf * 16);
            acc[t] = __builtin_amdgcn_wmma_f32_16x16x32_bf16(
                         false, A, false, Bv, (short)0, acc[t], false, false);
        }
        __syncthreads();                          // buffer reuse guard
    }

    #pragma unroll
    for (int t = 0; t < 4; ++t) {
        const int ncol = ng * 64 + t * 16 + (lane & 15);
        float bias;
        if (ncol < N_CLS) bias = bcls[ncol];
        else { int rr = ncol - N_CLS; if (rr > N_REG - 1) rr = N_REG - 1; bias = breg[rr]; }
        #pragma unroll
        for (int r = 0; r < 8; ++r) {
            const int mm = mt * 16 + khalf * 8 + r;
            head[(size_t)mm * N_HEAD_P + ncol] = acc[t][r] + bias;
        }
    }
}

// ---------------------------------------------------------------------------
// Assemble proposals (B, 714, 77): j<2 cls, j=2,3 pure anchor, j>=4 anchor+reg.
// anchors_anchor_dim is already in (p, j) proposal order.
// ---------------------------------------------------------------------------
__global__ void laneatt_assemble(const float* __restrict__ head,
                                 const float* __restrict__ anch_dim,
                                 float* __restrict__ props) {
    const long long total = (long long)B_BATCH * N_PROP * N_PRED;
    const long long gid = (long long)blockIdx.x * blockDim.x + threadIdx.x;
    if (gid >= total) return;
    const int j  = (int)(gid % N_PRED);
    const long long bp = gid / N_PRED;
    const int p  = (int)(bp % N_PROP);
    const int b  = (int)(bp / N_PROP);
    const int e  = p / N_ANG;
    const int a  = p - e * N_ANG;
    const int m  = b * N_EDGE + e;

    float v;
    if (j < 2)       v = head[(size_t)m * N_HEAD_P + a * 2 + j];
    else if (j < 4)  v = anch_dim[(size_t)p * N_PRED + j];
    else             v = anch_dim[(size_t)p * N_PRED + j]
                       + head[(size_t)m * N_HEAD_P + N_CLS + a * (N_OFF + 1) + (j - 4)];
    props[gid] = v;
}

// scores = softmax(logits)[...,1] = sigmoid(l1 - l0)
__global__ void laneatt_scores(const float* __restrict__ props,
                               float* __restrict__ scores) {
    const int n = B_BATCH * N_PROP;
    const int gid = blockIdx.x * blockDim.x + threadIdx.x;
    if (gid >= n) return;
    const float l0 = props[(size_t)gid * N_PRED + 0];
    const float l1 = props[(size_t)gid * N_PRED + 1];
    scores[gid] = 1.0f / (1.0f + __expf(l0 - l1));
}

// ---------------------------------------------------------------------------
// NMS: one workgroup per batch item.  714x72 xs table + metadata in LDS
// (~218 KB; CDNA5 allows 320 KB/workgroup).  xs row stride padded to 73
// floats (73 coprime with 64 banks -> conflict-free lane-parallel j access).
// ---------------------------------------------------------------------------
__global__ void laneatt_nms(const float* __restrict__ props,
                            const float* __restrict__ scores,
                            float* __restrict__ keep) {
    const int b   = blockIdx.x;
    const int tid = threadIdx.x;
    const int NT  = blockDim.x;

    extern __shared__ float smem[];
    float* xs     = smem;                          // 714 * 73
    float* stL    = xs  + N_PROP * 73;             // 714
    float* enL    = stL + N_PROP;                  // 714
    float* scL    = enL + N_PROP;                  // 714
    int*   orderL = (int*)(scL + N_PROP);          // 714
    int*   suppL  = orderL + N_PROP;               // 714

    for (int i = tid; i < N_PROP; i += NT) scL[i] = scores[(size_t)b * N_PROP + i];
    __syncthreads();

    // stable descending rank (matches jnp.argsort(-scores): ties -> lower idx)
    for (int i = tid; i < N_PROP; i += NT) {
        const float si = scL[i];
        int r = 0;
        for (int j = 0; j < N_PROP; ++j)
            r += (scL[j] > si) || (scL[j] == si && j < i);
        orderL[r] = i;
        suppL[i]  = 0;
    }
    __syncthreads();

    for (int i = tid; i < N_PROP; i += NT) {
        const float* pr = props + ((size_t)b * N_PROP + orderL[i]) * N_PRED;
        float st = rintf(pr[2] * (float)(N_OFF - 1));
        st = fminf(fmaxf(st, 0.0f), (float)(N_OFF - 1));
        float en = fminf(fmaxf(st + pr[4] - 1.0f, 0.0f), (float)(N_OFF - 1));
        stL[i] = st; enL[i] = en;
        float* xrow = &xs[(size_t)i * 73];
        for (int k = 0; k < N_OFF; ++k) xrow[k] = pr[5 + k];
    }
    __syncthreads();

    // sequential suppression scan (i uniform across block, j lane-parallel)
    for (int i = 0; i < N_PROP - 1; ++i) {
        __syncthreads();
        if (suppL[i]) continue;                    // block-uniform branch
        const float sti = stL[i], eni = enL[i];
        const float* xi = &xs[(size_t)i * 73];
        for (int j = i + 1 + tid; j < N_PROP; j += NT) {
            if (suppL[j]) continue;
            const float s   = fmaxf(sti, stL[j]);
            const float e   = fminf(eni, enL[j]);
            const float cnt = e - s + 1.0f;
            if (cnt > 0.0f) {
                int lo = (int)s;
                int hi = (int)floorf(e); if (hi > N_OFF - 1) hi = N_OFF - 1;
                const float* xj = &xs[(size_t)j * 73];
                float sum = 0.0f;
                for (int k = lo; k <= hi; ++k) sum += fabsf(xi[k] - xj[k]);
                if (sum / fmaxf(cnt, 1.0f) < NMS_THR) suppL[j] = 1;
            }
        }
    }
    __syncthreads();

    for (int i = tid; i < N_PROP; i += NT)
        keep[(size_t)b * N_PROP + orderL[i]] = suppL[i] ? 0.0f : 1.0f;
}

// ---------------------------------------------------------------------------
extern "C" void kernel_launch(void* const* d_in, const int* in_sizes, int n_in,
                              void* d_out, int out_size, void* d_ws, size_t ws_size,
                              hipStream_t stream) {
    const float* feat   = (const float*)d_in[0];
    const float* W_conv = (const float*)d_in[1];
    const float* b_conv = (const float*)d_in[2];
    const float* W_cls  = (const float*)d_in[3];
    const float* b_cls  = (const float*)d_in[4];
    const float* W_reg  = (const float*)d_in[5];
    const float* b_reg  = (const float*)d_in[6];
    // d_in[7] = anchors (unused; anchors_anchor_dim is in proposal order)
    const float* anch_d = (const float*)d_in[8];

    float* out = (float*)d_out;
    float* props  = out;                                           // 32*714*77
    float* keep   = out + (size_t)B_BATCH * N_PROP * N_PRED;       // 32*714
    float* scores = keep + (size_t)B_BATCH * N_PROP;               // 32*714

    // workspace layout (256B aligned): Wcb 1MB | Whb 2.5MB | featb 2.75MB | head 6.9MB
    char* ws = (char*)d_ws;
    size_t off = 0;
    auto take = [&](size_t bytes) { char* p = ws + off; off = (off + bytes + 255) & ~(size_t)255; return p; };
    __bf16* Wcb   = (__bf16*)take((size_t)C_MID * C_IN * sizeof(__bf16));
    __bf16* Whb   = (__bf16*)take((size_t)N_HEAD_P * C_MID * sizeof(__bf16));
    __bf16* featb = (__bf16*)take((size_t)M_ROWS * C_MID * sizeof(__bf16));
    float*  head  = (float*)take((size_t)M_ROWS * N_HEAD_P * sizeof(float));

    // 0) pack weights to bf16 (head weights fused + zero padded)
    {
        const int total = N_HEAD_P * C_MID + C_MID * C_IN;
        laneatt_pack_weights<<<(total + 255) / 256, 256, 0, stream>>>(
            W_conv, W_cls, W_reg, Wcb, Whb);
    }
    // 1) conv GEMM: 84 M-tiles x 16 col-groups = 1344 waves, 8 waves/block
    {
        const int waves = M_TILES * (C_MID / 64);
        laneatt_stage1_conv<<<(waves + 7) / 8, 256, 0, stream>>>(feat, Wcb, b_conv, featb);
    }
    // 2) head GEMM: grid (20 col-groups, 14 row-groups), 6 waves/block, TDM staging
    {
        dim3 grid(N_HEAD_P / 64, M_TILES / S2_WAVES);
        laneatt_stage2_heads<<<grid, 32 * S2_WAVES, 0, stream>>>(
            featb, Whb, b_cls, b_reg, head);
    }
    // 3) assemble proposals
    {
        const long long total = (long long)B_BATCH * N_PROP * N_PRED;
        laneatt_assemble<<<(unsigned)((total + 255) / 256), 256, 0, stream>>>(head, anch_d, props);
    }
    // 4) scores
    laneatt_scores<<<(B_BATCH * N_PROP + 255) / 256, 256, 0, stream>>>(props, scores);
    // 5) NMS: one block per batch, ~218 KB dynamic LDS
    {
        const size_t shmem = (size_t)(N_PROP * 73 + 3 * N_PROP) * sizeof(float)
                           + (size_t)(2 * N_PROP) * sizeof(int);
        laneatt_nms<<<B_BATCH, 256, shmem, stream>>>(props, scores, keep);
    }
}
